// MultiHeadAttention_87694642250090
// MI455X (gfx1250) — matrix operand order repaired and checked by reading, not compiled, not run on hardware
//
#include <hip/hip_runtime.h>
#include <hip/hip_bf16.h>

// Problem constants (match reference)
#define B_   2
#define T_   2048
#define D_   1024
#define H_   16
#define DK_  64
#define MROWS (B_ * T_)   // 4096

typedef __attribute__((ext_vector_type(16))) __bf16 v16bf;
typedef __attribute__((ext_vector_type(8)))  __bf16 v8bf;
typedef __attribute__((ext_vector_type(8)))  float  v8f;
typedef __attribute__((ext_vector_type(4)))  int    v4i;

// CDNA5 async global->LDS (ASYNCcnt-tracked), guarded so absence cannot break the build.
// Prototype (from clang diagnostic): (v4i addrspace(1)*, v4i addrspace(3)*, imm int, imm int)
#if __has_builtin(__builtin_amdgcn_global_load_async_to_lds_b128) && \
    __has_builtin(__builtin_amdgcn_s_wait_asynccnt)
#define HAVE_ASYNC_LDS 1
#define AS1V4(p) ((__attribute__((address_space(1))) v4i*)(unsigned long long)(p))
#define AS3V4(p) ((__attribute__((address_space(3))) v4i*)(unsigned)(unsigned long long)(p))
#else
#define HAVE_ASYNC_LDS 0
#endif

// float -> bf16 bits, round-to-nearest-even
static __device__ __forceinline__ unsigned short f2bf_bits(float f) {
    unsigned u = __builtin_bit_cast(unsigned, f);
    unsigned r = u + 0x7FFFu + ((u >> 16) & 1u);
    return (unsigned short)(r >> 16);
}
static __device__ __forceinline__ unsigned pack2(float a, float b) {
    return (unsigned)f2bf_bits(a) | ((unsigned)f2bf_bits(b) << 16);
}

// ---- WMMA fragment loaders ----------------------------------------------
// A-matrix 16x32 bf16 (ISA 7.12.2): lane 0-15 -> M=lane, lane 16-31 -> M, K+8.
// Per lane the 16 elements are two contiguous K-runs: [kl..kl+7] and [16+kl..16+kl+7]
// -> two 16B LDS loads (ds_load_b128 each).  Strides are multiples of 8 elements
// so every access below is 16-byte aligned.
static __device__ __forceinline__ v16bf load_a_frag(const unsigned short* As, int stride,
                                                    int row0, int kk, int lane) {
    const int row = row0 + (lane & 15);
    const int kl  = (lane >= 16) ? 8 : 0;
    const unsigned short* p = As + row * stride + kk + kl;
    v8bf lo = *(const v8bf*)(p);
    v8bf hi = *(const v8bf*)(p + 16);
    return __builtin_shufflevector(lo, hi, 0, 1, 2, 3, 4, 5, 6, 7,
                                           8, 9, 10, 11, 12, 13, 14, 15);
}

// B-matrix 32x16 bf16, fragment-ready (transposed) LDS layout BsT[n][k]:
// lane N = lane&15, elements e<8 -> K = kb + e, e>=8 -> K = kb + 8 + e  (kb = 0 / 8 for lane halves)
// -> two 16B runs = two ds_load_b128.
static __device__ __forceinline__ v16bf load_b_fragT(const unsigned short* BsT, int stride,
                                                     int col0, int kk, int lane) {
    const int col = col0 + (lane & 15);
    const int kb  = kk + ((lane >= 16) ? 8 : 0);
    const unsigned short* p = BsT + col * stride + kb;
    v8bf lo = *(const v8bf*)(p);
    v8bf hi = *(const v8bf*)(p + 16);
    return __builtin_shufflevector(lo, hi, 0, 1, 2, 3, 4, 5, 6, 7,
                                           8, 9, 10, 11, 12, 13, 14, 15);
}

static __device__ __forceinline__ v8f wmma_bf16(v16bf a, v16bf b, v8f c) {
    return __builtin_amdgcn_wmma_f32_16x16x32_bf16(false, a, false, b, (short)0, c, false, false);
}

// =====================================================================
// Kernel 1/6: dense GEMM  Y = X[M,K] * W[K,N] + bias
//   mode 0: Y = bf16, scattered head-major [B,H,T,DK]   (QKV projections)
//   mode 1: Y = fp32, row-major [M,N]                    (output projection)
// Tiles: 128x128x32, 256 threads (8 waves), wave = 32x64 block = 2x4 WMMA tiles
// =====================================================================
__global__ __launch_bounds__(256) void proj_gemm(const float* __restrict__ X,
                                                 const float* __restrict__ W,
                                                 const float* __restrict__ bias,
                                                 void* __restrict__ Yv,
                                                 int M, int N, int Kdim, int mode) {
    __shared__ __attribute__((aligned(16))) unsigned short As [128 * 40];  // [row][k], stride 40
    __shared__ __attribute__((aligned(16))) unsigned short BsT[128 * 40];  // [col][k], stride 40

    const int tid  = threadIdx.x;
    const int lane = tid & 31;
    const int wave = tid >> 5;
    const int waveRow = (wave >> 1) * 32;  // 0,32,64,96
    const int waveCol = (wave & 1) * 64;   // 0,64
    const int rowBase = blockIdx.y * 128;
    const int colBase = blockIdx.x * 128;

    v8f acc[2][4] = {};

    for (int k0 = 0; k0 < Kdim; k0 += 32) {
        if (k0 + 32 < Kdim) {   // CDNA5 global_prefetch_b8 on next K-slab
            __builtin_prefetch(&X[(size_t)(rowBase + (tid >> 1)) * Kdim + k0 + 32 + (tid & 1) * 16], 0, 1);
            __builtin_prefetch(&W[(size_t)(k0 + 32 + (tid >> 3)) * N + colBase + (tid & 7) * 16], 0, 1);
        }
        // A tile 128x32: float4 loads, packed 8B LDS stores
#pragma unroll
        for (int i = 0; i < 4; ++i) {
            int idx = i * 256 + tid;           // 1024 float4 chunks
            int r = idx >> 3, c4 = idx & 7;    // 8 float4 per 32-wide row
            float4 f = *(const float4*)(X + (size_t)(rowBase + r) * Kdim + k0 + c4 * 4);
            unsigned long long pk = (unsigned long long)pack2(f.x, f.y) |
                                    ((unsigned long long)pack2(f.z, f.w) << 32);
            *(unsigned long long*)(As + r * 40 + c4 * 4) = pk;
        }
        // B tile 32x128 -> transposed into BsT[col][k]
#pragma unroll
        for (int i = 0; i < 4; ++i) {
            int idx = i * 256 + tid;           // 1024 float4 chunks
            int r = idx >> 5, c4 = idx & 31;   // 32 float4 per 128-wide row
            float4 f = *(const float4*)(W + (size_t)(k0 + r) * N + colBase + c4 * 4);
            BsT[(c4 * 4 + 0) * 40 + r] = f2bf_bits(f.x);
            BsT[(c4 * 4 + 1) * 40 + r] = f2bf_bits(f.y);
            BsT[(c4 * 4 + 2) * 40 + r] = f2bf_bits(f.z);
            BsT[(c4 * 4 + 3) * 40 + r] = f2bf_bits(f.w);
        }
        __syncthreads();

        v16bf af[2], bf[4];
#pragma unroll
        for (int mi = 0; mi < 2; ++mi) af[mi] = load_a_frag(As, 40, waveRow + mi * 16, 0, lane);
#pragma unroll
        for (int ni = 0; ni < 4; ++ni) bf[ni] = load_b_fragT(BsT, 40, waveCol + ni * 16, 0, lane);
#pragma unroll
        for (int mi = 0; mi < 2; ++mi)
#pragma unroll
            for (int ni = 0; ni < 4; ++ni)
                acc[mi][ni] = wmma_bf16(af[mi], bf[ni], acc[mi][ni]);
        __syncthreads();
    }

    // Epilogue: bias + store.  C/D layout: row = r + 8*laneHi, col = lane&15.
#pragma unroll
    for (int mi = 0; mi < 2; ++mi)
#pragma unroll
        for (int ni = 0; ni < 4; ++ni) {
            const int col = colBase + waveCol + ni * 16 + (lane & 15);
            const float bv = bias[col];
#pragma unroll
            for (int r = 0; r < 8; ++r) {
                const int row = rowBase + waveRow + mi * 16 + r + ((lane >= 16) ? 8 : 0);
                const float vv = acc[mi][ni][r] + bv;
                if (mode == 0) {
                    int b = row >> 11, t = row & (T_ - 1);     // T_ = 2048
                    int h = col >> 6,  dk = col & (DK_ - 1);   // DK_ = 64
                    ((unsigned short*)Yv)[(((size_t)(b * H_ + h) * T_) + t) * DK_ + dk] = f2bf_bits(vv);
                } else {
                    ((float*)Yv)[(size_t)row * N + col] = vv;
                }
            }
        }
}

// =====================================================================
// Kernel 2/6: scores = scale * Q Kt with causal mask, per (b,h)
// Q,K bf16 [T,DK]; S fp32 [T,T].  128x128 tile, K-dim = 64 (no outer loop).
// Fragment-ready B layout == K's own row-major layout -> tile fill is pure
// 16B copies -> async global->LDS (GLOBAL_LOAD_ASYNC_TO_LDS_B128) when available.
// =====================================================================
__global__ __launch_bounds__(256) void scores_kernel(const unsigned short* __restrict__ Qb,
                                                     const unsigned short* __restrict__ Kb,
                                                     float* __restrict__ attn) {
    const int bh = blockIdx.z;
    const unsigned short* Q = Qb + (size_t)bh * T_ * DK_;
    const unsigned short* K = Kb + (size_t)bh * T_ * DK_;
    float* S = attn + (size_t)bh * T_ * T_;

    const int tid  = threadIdx.x;
    const int rowBase = blockIdx.y * 128;
    const int colBase = blockIdx.x * 128;

    if (colBase > rowBase) {   // fully-masked block: emit -inf (softmax -> 0)
        const float ninf = -__builtin_inff();
        const float4 nf4 = make_float4(ninf, ninf, ninf, ninf);
#pragma unroll
        for (int i = 0; i < 16; ++i) {         // 4096 float4 chunks
            int idx = i * 256 + tid;
            int r = idx >> 5, c4 = idx & 31;   // 32 float4 per 128-wide row
            *(float4*)(S + (size_t)(rowBase + r) * T_ + colBase + c4 * 4) = nf4;
        }
        return;
    }

    __shared__ __attribute__((aligned(16))) unsigned short Asd [128 * 72];  // Q  [row][k], stride 72
    __shared__ __attribute__((aligned(16))) unsigned short BsdT[128 * 72];  // Kt [col][k], stride 72

    const int lane = tid & 31;
    const int wave = tid >> 5;
    const int waveRow = (wave >> 1) * 32;
    const int waveCol = (wave & 1) * 64;

#if HAVE_ASYNC_LDS
#pragma unroll
    for (int i = 0; i < 4; ++i) {              // 1024 16B chunks each
        int idx = i * 256 + tid;
        int r = idx >> 3, c8 = idx & 7;        // 8 chunks per 64-wide row
        __builtin_amdgcn_global_load_async_to_lds_b128(
            AS1V4(Q + (size_t)(rowBase + r) * DK_ + c8 * 8),
            AS3V4(Asd + r * 72 + c8 * 8), 0, 0);
        __builtin_amdgcn_global_load_async_to_lds_b128(
            AS1V4(K + (size_t)(colBase + r) * DK_ + c8 * 8),
            AS3V4(BsdT + r * 72 + c8 * 8), 0, 0);
    }
    __builtin_amdgcn_s_wait_asynccnt(0);
#else
#pragma unroll
    for (int i = 0; i < 4; ++i) {              // 1024 uint4 chunks each
        int idx = i * 256 + tid;
        int r = idx >> 3, c8 = idx & 7;        // 8 chunks per 64-wide row
        *(uint4*)(Asd + r * 72 + c8 * 8) =
            *(const uint4*)(Q + (size_t)(rowBase + r) * DK_ + c8 * 8);
        *(uint4*)(BsdT + r * 72 + c8 * 8) =
            *(const uint4*)(K + (size_t)(colBase + r) * DK_ + c8 * 8);
    }
#endif
    __syncthreads();

    v8f acc[2][4] = {};
#pragma unroll
    for (int kk = 0; kk < 64; kk += 32) {
        v16bf af[2], bf[4];
#pragma unroll
        for (int mi = 0; mi < 2; ++mi) af[mi] = load_a_frag(Asd, 72, waveRow + mi * 16, kk, lane);
#pragma unroll
        for (int ni = 0; ni < 4; ++ni) bf[ni] = load_b_fragT(BsdT, 72, waveCol + ni * 16, kk, lane);
#pragma unroll
        for (int mi = 0; mi < 2; ++mi)
#pragma unroll
            for (int ni = 0; ni < 4; ++ni)
                acc[mi][ni] = wmma_bf16(af[mi], bf[ni], acc[mi][ni]);
    }

    const float scale = 0.125f;               // 1/sqrt(64)
    const float ninf  = -__builtin_inff();
#pragma unroll
    for (int mi = 0; mi < 2; ++mi)
#pragma unroll
        for (int ni = 0; ni < 4; ++ni) {
            const int col = colBase + waveCol + ni * 16 + (lane & 15);
#pragma unroll
            for (int r = 0; r < 8; ++r) {
                const int row = rowBase + waveRow + mi * 16 + r + ((lane >= 16) ? 8 : 0);
                S[(size_t)row * T_ + col] = (col > row) ? ninf : acc[mi][ni][r] * scale;
            }
        }
}

// =====================================================================
// Kernel 3/6: row softmax in place. 1 block (256 thr) per row of 2048.
// =====================================================================
__global__ __launch_bounds__(256) void softmax_kernel(float* __restrict__ attn) {
    float* p = attn + (size_t)blockIdx.x * T_;
    const int tid = threadIdx.x;
    __shared__ float red[8];

    float4 va = *(const float4*)(p + tid * 4);
    float4 vb = *(const float4*)(p + 1024 + tid * 4);
    float lmax = fmaxf(fmaxf(fmaxf(va.x, va.y), fmaxf(va.z, va.w)),
                       fmaxf(fmaxf(vb.x, vb.y), fmaxf(vb.z, vb.w)));
#pragma unroll
    for (int off = 16; off > 0; off >>= 1) lmax = fmaxf(lmax, __shfl_down(lmax, off, 32));
    if ((tid & 31) == 0) red[tid >> 5] = lmax;
    __syncthreads();
    float bmax = red[0];
#pragma unroll
    for (int i = 1; i < 8; ++i) bmax = fmaxf(bmax, red[i]);

    va.x = __expf(va.x - bmax); va.y = __expf(va.y - bmax);
    va.z = __expf(va.z - bmax); va.w = __expf(va.w - bmax);
    vb.x = __expf(vb.x - bmax); vb.y = __expf(vb.y - bmax);
    vb.z = __expf(vb.z - bmax); vb.w = __expf(vb.w - bmax);
    float lsum = (va.x + va.y + va.z + va.w) + (vb.x + vb.y + vb.z + vb.w);
#pragma unroll
    for (int off = 16; off > 0; off >>= 1) lsum += __shfl_down(lsum, off, 32);
    __syncthreads();
    if ((tid & 31) == 0) red[tid >> 5] = lsum;
    __syncthreads();
    float bsum = 0.f;
#pragma unroll
    for (int i = 0; i < 8; ++i) bsum += red[i];

    const float inv = 1.0f / bsum;   // bsum >= 1 (diagonal term)
    va.x *= inv; va.y *= inv; va.z *= inv; va.w *= inv;
    vb.x *= inv; vb.y *= inv; vb.z *= inv; vb.w *= inv;
    *(float4*)(p + tid * 4) = va;
    *(float4*)(p + 1024 + tid * 4) = vb;
}

// =====================================================================
// Kernel 4/6: ctx = attn[T,T] * V[T,DK] per (b,h); causal truncation of K-loop.
// 128x64 block, 8 waves, wave = 16 rows x 64 cols (1x4 WMMA tiles).
// Output fp32, re-merged heads: ctx[(b*T+row)*D + h*DK + col]
// =====================================================================
__global__ __launch_bounds__(256) void ctx_kernel(const float* __restrict__ attn,
                                                  const unsigned short* __restrict__ Vb,
                                                  float* __restrict__ ctx) {
    const int bh = blockIdx.z;
    const int b  = bh / H_, h = bh - b * H_;
    const float* A = attn + (size_t)bh * T_ * T_;
    const unsigned short* V = Vb + (size_t)bh * T_ * DK_;

    const int tid  = threadIdx.x;
    const int lane = tid & 31;
    const int wave = tid >> 5;
    const int rowBase = blockIdx.y * 128;

    __shared__ __attribute__((aligned(16))) unsigned short Asd[128 * 40];  // attn [row][k], stride 40
    __shared__ __attribute__((aligned(16))) unsigned short BsT[64 * 40];   // Vt   [col][k], stride 40

    v8f acc[4] = {};
    const int kEnd = rowBase + 128;           // attn[row][j]==0 for j>row

    for (int k0 = 0; k0 < kEnd; k0 += 32) {
        // attn tile 128x32: float4 load, fp32 -> bf16, packed 8B LDS store
#pragma unroll
        for (int i = 0; i < 4; ++i) {
            int idx = i * 256 + tid;           // 1024 float4 chunks
            int r = idx >> 3, c4 = idx & 7;
            float4 f = *(const float4*)(A + (size_t)(rowBase + r) * T_ + k0 + c4 * 4);
            unsigned long long pk = (unsigned long long)pack2(f.x, f.y) |
                                    ((unsigned long long)pack2(f.z, f.w) << 32);
            *(unsigned long long*)(Asd + r * 40 + c4 * 4) = pk;
        }
        // V tile 32x64 -> transposed BsT[col][k]; uint4 source reads
        {
            int k = tid >> 3, n8 = tid & 7;    // 256 chunks, one per thread
            union { uint4 v; unsigned short s[8]; } u;
            u.v = *(const uint4*)(V + (size_t)(k0 + k) * DK_ + n8 * 8);
#pragma unroll
            for (int j = 0; j < 8; ++j)
                BsT[(n8 * 8 + j) * 40 + k] = u.s[j];
        }
        __syncthreads();

        v16bf af = load_a_frag(Asd, 40, wave * 16, 0, lane);
#pragma unroll
        for (int ni = 0; ni < 4; ++ni) {
            v16bf bf = load_b_fragT(BsT, 40, ni * 16, 0, lane);
            acc[ni] = wmma_bf16(af, bf, acc[ni]);
        }
        __syncthreads();
    }

#pragma unroll
    for (int ni = 0; ni < 4; ++ni) {
        const int col = ni * 16 + (lane & 15);        // dk within head
#pragma unroll
        for (int r = 0; r < 8; ++r) {
            const int row = rowBase + wave * 16 + r + ((lane >= 16) ? 8 : 0);
            ctx[((size_t)b * T_ + row) * D_ + h * DK_ + col] = acc[ni][r];
        }
    }
}

// =====================================================================
extern "C" void kernel_launch(void* const* d_in, const int* in_sizes, int n_in,
                              void* d_out, int out_size, void* d_ws, size_t ws_size,
                              hipStream_t stream) {
    (void)in_sizes; (void)n_in; (void)out_size; (void)ws_size;

    const float* q_inp = (const float*)d_in[0];
    const float* k_inp = (const float*)d_in[1];
    const float* v_inp = (const float*)d_in[2];
    // d_in[3] = attn_mask (bool causal) — reproduced analytically
    const float* Wq = (const float*)d_in[4];
    const float* bq = (const float*)d_in[5];
    const float* Wk = (const float*)d_in[6];
    const float* bk = (const float*)d_in[7];
    const float* Wv = (const float*)d_in[8];
    const float* bv = (const float*)d_in[9];
    const float* Wo = (const float*)d_in[10];
    const float* bo = (const float*)d_in[11];

    float* out  = (float*)d_out;                               // [B,T,D]
    float* attn = out + (size_t)B_ * T_ * D_;                  // [B,H,T,T]

    // workspace: Q/K/V bf16 (8 MB each) + ctx fp32 (16 MB) = 40 MB
    unsigned short* Qb = (unsigned short*)d_ws;
    unsigned short* Kb = Qb + (size_t)MROWS * D_;
    unsigned short* Vb = Kb + (size_t)MROWS * D_;
    float* ctx = (float*)(Vb + (size_t)MROWS * D_);

    dim3 blk(256);
    dim3 gproj(D_ / 128, MROWS / 128);                         // (8, 32)

    proj_gemm<<<gproj, blk, 0, stream>>>(q_inp, Wq, bq, Qb, MROWS, D_, D_, 0);
    proj_gemm<<<gproj, blk, 0, stream>>>(k_inp, Wk, bk, Kb, MROWS, D_, D_, 0);
    proj_gemm<<<gproj, blk, 0, stream>>>(v_inp, Wv, bv, Vb, MROWS, D_, D_, 0);

    dim3 gsc(T_ / 128, T_ / 128, B_ * H_);                     // (16,16,32)
    scores_kernel<<<gsc, blk, 0, stream>>>(Qb, Kb, attn);

    softmax_kernel<<<dim3(B_ * H_ * T_), blk, 0, stream>>>(attn);

    ctx_kernel<<<dim3(1, T_ / 128, B_ * H_), blk, 0, stream>>>(attn, Vb, ctx);

    proj_gemm<<<gproj, blk, 0, stream>>>(ctx, Wo, bo, out, MROWS, D_, D_, 1);
}
